// surrogate_GCN_49400713838982
// MI455X (gfx1250) — matrix operand order repaired and checked by reading, not compiled, not run on hardware
//
#include <hip/hip_runtime.h>
#include <stdint.h>

typedef __attribute__((ext_vector_type(16))) __bf16 v16bf;
typedef __attribute__((ext_vector_type(8)))  __bf16 v8bf;
typedef __attribute__((ext_vector_type(4)))  __bf16 v4bf;
typedef __attribute__((ext_vector_type(8)))  float  v8f;

// round-to-nearest-even f32 -> bf16 via integer math
__device__ __forceinline__ __bf16 f2bf(float f) {
    uint32_t u = __builtin_bit_cast(uint32_t, f);
    u += 0x7FFFu + ((u >> 16) & 1u);
    return __builtin_bit_cast(__bf16, (uint16_t)(u >> 16));
}

// low 32 bits of a flat pointer into __shared__ = LDS byte offset
// (ISA: LDS flat aperture occupies addr[63:32])
__device__ __forceinline__ uint32_t lds_addr32(const void* p) {
    return (uint32_t)(uintptr_t)p;
}

// ---------------- degree / normalization ----------------

__global__ void k_init_deg(uint32_t* __restrict__ deg, int n) {
    int v = blockIdx.x * blockDim.x + threadIdx.x;
    if (v < n) deg[v] = 1u;                      // self-loop
}

__global__ void k_count_deg(const long long* __restrict__ dst,
                            uint32_t* __restrict__ deg, int e) {
    int i = blockIdx.x * blockDim.x + threadIdx.x;
    if (i < e) atomicAdd(&deg[(int)dst[i]], 1u);
}

__global__ void k_dinv(const uint32_t* __restrict__ deg,
                       float* __restrict__ dinv, int n) {
    int v = blockIdx.x * blockDim.x + threadIdx.x;
    if (v < n) dinv[v] = rsqrtf((float)deg[v]);  // deg >= 1 always
}

// ---------------- one-time bf16 conversions ----------------

// X (N x 128 f32) -> Xb (N x 128 bf16), 8 elements per thread
__global__ void k_cvt_x(const float* __restrict__ X, __bf16* __restrict__ Xb,
                        int total8) {
    int idx = blockIdx.x * blockDim.x + threadIdx.x;
    if (idx >= total8) return;
    const float4 a = *(const float4*)(X + (size_t)idx * 8);
    const float4 b = *(const float4*)(X + (size_t)idx * 8 + 4);
    v8bf o;
    o[0] = f2bf(a.x); o[1] = f2bf(a.y); o[2] = f2bf(a.z); o[3] = f2bf(a.w);
    o[4] = f2bf(b.x); o[5] = f2bf(b.y); o[6] = f2bf(b.z); o[7] = f2bf(b.w);
    *(v8bf*)(Xb + (size_t)idx * 8) = o;
}

// W1 (128 x 64) -> W1t bf16 [64][128]  (column-major => contiguous per-N rows)
__global__ void k_cvt_w1(const float* __restrict__ W1, __bf16* __restrict__ W1t) {
    int t = blockIdx.x * blockDim.x + threadIdx.x;   // 64*128
    if (t >= 64 * 128) return;
    int n = t >> 7, k = t & 127;
    W1t[n * 128 + k] = f2bf(W1[(size_t)k * 64 + n]);
}

// W2 (64 x 40) -> W2t bf16 [48][64], columns 40..47 zero-padded
__global__ void k_cvt_w2(const float* __restrict__ W2, __bf16* __restrict__ W2t) {
    int t = blockIdx.x * blockDim.x + threadIdx.x;   // 48*64
    if (t >= 48 * 64) return;
    int n = t >> 6, k = t & 63;
    W2t[n * 64 + k] = (n < 40) ? f2bf(W2[(size_t)k * 40 + n]) : f2bf(0.0f);
}

// ---------------- GEMM1: H = Xb(100k x128 bf16) @ W1, bf16 WMMA ----------------
// block = 128 threads = 4 waves; A strip (16x128 bf16, 4KB) async-staged to LDS
// once per block; wave w computes the 16x16 tile at (blockIdx*16, w*16).
// Epilogue seeds A1 with the self-loop term: A1 = H * dinv^2.

__global__ void __launch_bounds__(128)
k_gemm1(const __bf16* __restrict__ Xb, const __bf16* __restrict__ W1t,
        const float* __restrict__ dinv, float* __restrict__ H,
        float* __restrict__ A1) {
    __shared__ __bf16 atile[16 * 128];               // 4096 bytes

    const int wave = threadIdx.x >> 5;
    const int lane = threadIdx.x & 31;
    const int mloc = lane & 15;
    const int hi   = lane >> 4;
    const int m0   = blockIdx.x * 16;
    const int ncol = wave * 16 + mloc;

    // ---- async stage: 128 threads x 32B = 4KB (two b128 issues, shared addr) ----
    {
        uint32_t la = lds_addr32(&atile[threadIdx.x * 16]);
        uint64_t ga = (uint64_t)(uintptr_t)(Xb + (size_t)m0 * 128 + threadIdx.x * 16);
        asm volatile("global_load_async_to_lds_b128 %0, %1, off"
                     :: "v"(la), "v"(ga) : "memory");
        asm volatile("global_load_async_to_lds_b128 %0, %1, off offset:16"
                     :: "v"(la), "v"(ga) : "memory");
        asm volatile("s_wait_asynccnt 0" ::: "memory");
    }
    __syncthreads();

    const __bf16* As = &atile[mloc * 128 + 16 * hi]; // per-lane A row base
    const __bf16* Bs = W1t + (size_t)ncol * 128;     // per-lane B column base

    v8f c = {};
#pragma unroll
    for (int kb = 0; kb < 128; kb += 32) {
        v8bf alo = *(const v8bf*)(As + kb);          // K = kb+16*hi   .. +8
        v8bf ahi = *(const v8bf*)(As + kb + 8);      // K = kb+16*hi+8 .. +8
        v16bf a = __builtin_shufflevector(alo, ahi,
                     0, 1, 2, 3, 4, 5, 6, 7, 8, 9, 10, 11, 12, 13, 14, 15);
        v16bf b = *(const v16bf*)(Bs + kb + 16 * hi);
        c = __builtin_amdgcn_wmma_f32_16x16x32_bf16(false, a, false, b,
                                                    (short)0, c, false, false);
    }
#pragma unroll
    for (int r = 0; r < 8; ++r) {
        int m = m0 + r + 8 * hi;
        float val = c[r];
        float di  = dinv[m];
        H[(size_t)m * 64 + ncol]  = val;
        A1[(size_t)m * 64 + ncol] = val * di * di;   // self-loop init
    }
}

// ---------------- edge scatter, layer 1: 16 threads/edge, float4 each ----------------

__global__ void k_scatter1(const long long* __restrict__ src,
                           const long long* __restrict__ dst,
                           const float* __restrict__ dinv,
                           const float* __restrict__ H,
                           float* __restrict__ A1, int e) {
    long long idx = (long long)blockIdx.x * blockDim.x + threadIdx.x;
    if (idx >= (long long)e * 16) return;
    int ei = (int)(idx >> 4);
    int q  = (int)(idx & 15);
    int s = (int)src[ei];
    int d = (int)dst[ei];
    float w = dinv[s] * dinv[d];
    const float4 h = *(const float4*)(H + (size_t)s * 64 + q * 4);
    float* o = A1 + (size_t)d * 64 + q * 4;
    atomicAdd(o + 0, h.x * w);
    atomicAdd(o + 1, h.y * w);
    atomicAdd(o + 2, h.z * w);
    atomicAdd(o + 3, h.w * w);
}

// ---------------- bias + relu + convert to bf16 (input of GEMM2) ----------------

__global__ void k_relu_cvt(const float* __restrict__ A1,
                           const float* __restrict__ b1,
                           __bf16* __restrict__ H1, int n) {
    int idx = blockIdx.x * blockDim.x + threadIdx.x;  // n*16 tasks, 4 feats each
    if (idx >= n * 16) return;
    int v = idx >> 4, q = idx & 15;
    const float4 a  = *(const float4*)(A1 + (size_t)v * 64 + q * 4);
    const float4 bb = *(const float4*)(b1 + q * 4);
    v4bf o;
    o[0] = f2bf(fmaxf(a.x + bb.x, 0.0f));
    o[1] = f2bf(fmaxf(a.y + bb.y, 0.0f));
    o[2] = f2bf(fmaxf(a.z + bb.z, 0.0f));
    o[3] = f2bf(fmaxf(a.w + bb.w, 0.0f));
    *(v4bf*)(H1 + (size_t)v * 64 + q * 4) = o;
}

// ---------------- GEMM2: H2 = H1(100k x64 bf16) @ W2, N padded to 48 ----------------
// block = 128 threads: all 4 waves stage the 16x64 bf16 A strip (2KB) via async
// copy; waves 0..2 compute N-tiles 0..2 (uniform wave-level branch, EXEC all-1s
// inside each computing wave). Epilogue seeds OUT = b2 + H2 * dinv^2.

__global__ void __launch_bounds__(128)
k_gemm2(const __bf16* __restrict__ H1, const __bf16* __restrict__ W2t,
        const float* __restrict__ b2, const float* __restrict__ dinv,
        float* __restrict__ H2, float* __restrict__ OUT) {
    __shared__ __bf16 atile[16 * 64];                // 2048 bytes

    const int wave = threadIdx.x >> 5;
    const int lane = threadIdx.x & 31;
    const int mloc = lane & 15;
    const int hi   = lane >> 4;
    const int m0   = blockIdx.x * 16;
    const int ncol = wave * 16 + mloc;

    // ---- async stage: 128 threads x 16B = 2KB ----
    {
        uint32_t la = lds_addr32(&atile[threadIdx.x * 8]);
        uint64_t ga = (uint64_t)(uintptr_t)(H1 + (size_t)m0 * 64 + threadIdx.x * 8);
        asm volatile("global_load_async_to_lds_b128 %0, %1, off"
                     :: "v"(la), "v"(ga) : "memory");
        asm volatile("s_wait_asynccnt 0" ::: "memory");
    }
    __syncthreads();

    if (wave >= 3) return;                           // uniform: wave 3 only staged

    const __bf16* As = &atile[mloc * 64 + 16 * hi];
    const __bf16* Bs = W2t + (size_t)ncol * 64;

    v8f c = {};
#pragma unroll
    for (int kb = 0; kb < 64; kb += 32) {
        v8bf alo = *(const v8bf*)(As + kb);
        v8bf ahi = *(const v8bf*)(As + kb + 8);
        v16bf a = __builtin_shufflevector(alo, ahi,
                     0, 1, 2, 3, 4, 5, 6, 7, 8, 9, 10, 11, 12, 13, 14, 15);
        v16bf b = *(const v16bf*)(Bs + kb + 16 * hi);
        c = __builtin_amdgcn_wmma_f32_16x16x32_bf16(false, a, false, b,
                                                    (short)0, c, false, false);
    }
    if (ncol < 40) {   // divergence only after all WMMAs
#pragma unroll
        for (int r = 0; r < 8; ++r) {
            int m = m0 + r + 8 * hi;
            float val = c[r];
            float di  = dinv[m];
            H2[(size_t)m * 40 + ncol]  = val;
            OUT[(size_t)m * 40 + ncol] = b2[ncol] + val * di * di;
        }
    }
}

// ---------------- edge scatter, layer 2: 10 float4-quads per edge ----------------

__global__ void k_scatter2(const long long* __restrict__ src,
                           const long long* __restrict__ dst,
                           const float* __restrict__ dinv,
                           const float* __restrict__ H2,
                           float* __restrict__ OUT, int e) {
    unsigned idx = blockIdx.x * blockDim.x + threadIdx.x;
    if (idx >= (unsigned)e * 10u) return;
    unsigned ei = idx / 10u;
    unsigned q  = idx - ei * 10u;
    int s = (int)src[ei];
    int d = (int)dst[ei];
    float w = dinv[s] * dinv[d];
    const float4 h = *(const float4*)(H2 + (size_t)s * 40 + q * 4);
    float* o = OUT + (size_t)d * 40 + q * 4;
    atomicAdd(o + 0, h.x * w);
    atomicAdd(o + 1, h.y * w);
    atomicAdd(o + 2, h.z * w);
    atomicAdd(o + 3, h.w * w);
}

// ---------------- host launcher ----------------

static inline size_t alignup(size_t x) { return (x + 255) & ~(size_t)255; }

extern "C" void kernel_launch(void* const* d_in, const int* in_sizes, int n_in,
                              void* d_out, int out_size, void* d_ws, size_t ws_size,
                              hipStream_t stream) {
    const float*     X  = (const float*)d_in[0];
    const long long* EI = (const long long*)d_in[1];
    const float*     W1 = (const float*)d_in[2];
    const float*     b1 = (const float*)d_in[3];
    const float*     W2 = (const float*)d_in[4];
    const float*     b2 = (const float*)d_in[5];

    const int N = in_sizes[0] / 128;     // 100000 (divisible by 16)
    const int E = in_sizes[1] / 2;       // 1600000
    const long long* src = EI;
    const long long* dst = EI + E;

    char* ws = (char*)d_ws;
    size_t o_deg  = 0;
    size_t o_dinv = o_deg  + alignup((size_t)N * 4);
    size_t o_Xb   = o_dinv + alignup((size_t)N * 4);
    size_t o_W1t  = o_Xb   + alignup((size_t)N * 128 * 2);
    size_t o_W2t  = o_W1t  + alignup((size_t)64 * 128 * 2);
    size_t o_H    = o_W2t  + alignup((size_t)48 * 64 * 2);
    size_t o_A1   = o_H    + alignup((size_t)N * 64 * 4);

    uint32_t* deg  = (uint32_t*)(ws + o_deg);
    float*    dinv = (float*)   (ws + o_dinv);
    __bf16*   Xb   = (__bf16*)  (ws + o_Xb);
    __bf16*   W1t  = (__bf16*)  (ws + o_W1t);
    __bf16*   W2t  = (__bf16*)  (ws + o_W2t);
    float*    H    = (float*)   (ws + o_H);
    float*    A1   = (float*)   (ws + o_A1);
    __bf16*   H1   = (__bf16*)  (ws + o_H);    // overlays H (H dead after scatter1)
    float*    H2   = (float*)   (ws + o_A1);   // overlays A1 (A1 dead after relu)
    float*    OUT  = (float*)d_out;

    k_init_deg <<<(N + 255) / 256, 256, 0, stream>>>(deg, N);
    k_count_deg<<<(E + 255) / 256, 256, 0, stream>>>(dst, deg, E);
    k_dinv     <<<(N + 255) / 256, 256, 0, stream>>>(deg, dinv, N);

    k_cvt_x    <<<(N * 16 + 255) / 256, 256, 0, stream>>>(X, Xb, N * 16);
    k_cvt_w1   <<<(64 * 128 + 255) / 256, 256, 0, stream>>>(W1, W1t);
    k_cvt_w2   <<<(48 * 64 + 255) / 256, 256, 0, stream>>>(W2, W2t);

    k_gemm1    <<<N / 16, 128, 0, stream>>>(Xb, W1t, dinv, H, A1);

    long long t1 = (long long)E * 16;
    k_scatter1 <<<(int)((t1 + 255) / 256), 256, 0, stream>>>(src, dst, dinv, H, A1, E);

    k_relu_cvt <<<(N * 16 + 255) / 256, 256, 0, stream>>>(A1, b1, H1, N);

    k_gemm2    <<<N / 16, 128, 0, stream>>>(H1, W2t, b2, dinv, H2, OUT);

    long long t2 = (long long)E * 10;
    k_scatter2 <<<(int)((t2 + 255) / 256), 256, 0, stream>>>(src, dst, dinv, H2, OUT, E);
}
